// ConviSTFT_58849641889878
// MI455X (gfx1250) — matrix operand order repaired and checked by reading, not compiled, not run on hardware
//
#include <hip/hip_runtime.h>

typedef _Float16 v16h __attribute__((ext_vector_type(16)));
typedef float    v8f  __attribute__((ext_vector_type(8)));

#define BB     16
#define FF     257
#define TT     2000
#define C2F    514        // 2*F reduction length
#define KSTEPS 17         // ceil(514/32)
#define NWIN   400        // win_len
#define STRIDE 100
#define SLICE  300        // win_len - stride  (samples trimmed from front)
#define OUTL   200000     // (T-1)*stride + win_len - SLICE
#define BROW   48         // padded B row length in halfs (96B => 32B-aligned frags)

__global__ void istft_zero_kernel(float* __restrict__ out, int n) {
  int i = blockIdx.x * blockDim.x + threadIdx.x;
  if (i < n) out[i] = 0.0f;
}

// One block = one (b, t0..t0+15) tile of the frames GEMM, fused overlap-add.
__global__ __launch_bounds__(256)
void istft_wmma_kernel(const float* __restrict__ mag,
                       const float* __restrict__ phase,
                       const float* __restrict__ weight,
                       float* __restrict__ out) {
  // A fragment stored directly in WMMA per-lane layout: [lane][elem] (16 f16/lane)
  __shared__ _Float16 Asl[32 * 16];
  // B slab: 32 k-values x 400 n, stored [n][k] with padded row (32B aligned frags)
  __shared__ _Float16 Bsl[NWIN * BROW];

  const int tid  = threadIdx.x;
  const int lane = tid & 31;
  // wave index is physically uniform per wave; force it into an SGPR so the
  // nt<25 guards become scalar branches (EXEC stays all-ones around WMMA).
  const int wave = __builtin_amdgcn_readfirstlane(tid >> 5);   // 0..7
  const int half = lane >> 4;      // 0 or 1
  const int nl   = lane & 15;

  const int blk = blockIdx.x;              // 0..1999
  const int b   = blk / (TT / 16);
  const int t0  = (blk % (TT / 16)) * 16;

  v8f acc0 = {}, acc1 = {}, acc2 = {}, acc3 = {};

  for (int kk = 0; kk < KSTEPS; ++kk) {
    const int c0 = kk * 32;

    // ---- A tile: 16 t-rows x 32 c-cols, 2 elements per thread ----
    {
      const int m = tid & 15;                     // t-row within tile
#pragma unroll
      for (int e = 0; e < 2; ++e) {
        const int k = (tid >> 4) + e * 16;        // 0..31 within k-step
        const int c = c0 + k;
        float val = 0.0f;
        if (c < C2F) {
          const int f = (c < FF) ? c : (c - FF);
          const long idx = ((long)(b * FF + f)) * TT + (t0 + m);
          const float mg = mag[idx];
          const float ph = phase[idx];
          val = mg * ((c < FF) ? __cosf(ph) : __sinf(ph));
        }
        // map (m,k) -> A layout matching B: lane = m + 16*(k/16), element = k%16
        const int fl = m + (k >> 4) * 16;
        const int fi = k & 15;
        Asl[fl * 16 + fi] = (_Float16)val;
      }
    }

    // ---- B slab: 32 k x 400 n, packed k-pairs (dword LDS stores) ----
#pragma unroll 5
    for (int p = tid; p < 16 * NWIN; p += 256) {
      const int n  = p % NWIN;                    // coalesced in n across lanes
      const int k  = (p / NWIN) * 2;              // even k
      union { _Float16 h[2]; unsigned u; } pk;
#pragma unroll
      for (int e = 0; e < 2; ++e) {
        const int c = c0 + k + e;
        pk.h[e] = (_Float16)((c < C2F) ? weight[c * NWIN + n] : 0.0f);
      }
      *(unsigned*)&Bsl[n * BROW + k] = pk.u;      // byte off = n*96 + 2k (4B aligned)
    }

    __syncthreads();

    // ---- WMMA: A shared by all waves; wave w covers n-tiles w, w+8, w+16(,24)
    const v16h afrag = *(const v16h*)&Asl[lane * 16];        // 32B aligned
#pragma unroll
    for (int s = 0; s < 4; ++s) {
      const int nt = wave + s * 8;               // scalar (SGPR) value
      if (nt < 25) {                             // scalar branch, EXEC untouched
        const int n = nt * 16 + nl;
        const v16h bfrag = *(const v16h*)&Bsl[n * BROW + half * 16]; // 32B aligned
        v8f* accp = (s == 0) ? &acc0 : (s == 1) ? &acc1 : (s == 2) ? &acc2 : &acc3;
        *accp = __builtin_amdgcn_wmma_f32_16x16x32_f16(
            false, afrag, false, bfrag, (short)0, *accp, false, false);
      }
    }

    __syncthreads();   // protect LDS before next fill
  }

  // ---- fused overlap-add: D layout lanes0-15 -> M=r, lanes16-31 -> M=8+r
#pragma unroll
  for (int s = 0; s < 4; ++s) {
    const int nt = wave + s * 8;
    if (nt < 25) {
      const v8f acc = (s == 0) ? acc0 : (s == 1) ? acc1 : (s == 2) ? acc2 : acc3;
      const int n = nt * 16 + nl;
#pragma unroll
      for (int r = 0; r < 8; ++r) {
        const int m = half * 8 + r;
        const int t = t0 + m;
        const int l = t * STRIDE + n - SLICE;
        if (l >= 0 && l < OUTL) {
          atomicAdd(&out[(long)b * OUTL + l], acc[r]);
        }
      }
    }
  }
}

// Divide by overlap-added window^2 (input-independent, <=4 terms per sample).
__global__ void istft_norm_kernel(float* __restrict__ out,
                                  const float* __restrict__ window) {
  const long i = (long)blockIdx.x * blockDim.x + threadIdx.x;
  if (i >= (long)BB * OUTL) return;
  const int j = (int)(i % OUTL);
  const int l = j + SLICE;                 // position in un-sliced signal
  const int tmin = j / STRIDE;             // floor((l-300)/100) >= 0
  int tmax = l / STRIDE;
  if (tmax > TT - 1) tmax = TT - 1;
  float coff = 0.0f;
  for (int t = tmin; t <= tmax; ++t) {     // at most 4 iterations
    const float w = window[l - t * STRIDE];
    coff += w * w;
  }
  out[i] = out[i] / (coff + 1e-12f);
}

extern "C" void kernel_launch(void* const* d_in, const int* in_sizes, int n_in,
                              void* d_out, int out_size, void* d_ws, size_t ws_size,
                              hipStream_t stream) {
  const float* mag    = (const float*)d_in[0];   // [16,257,2000]
  const float* phase  = (const float*)d_in[1];   // [16,257,2000]
  const float* weight = (const float*)d_in[2];   // [514,400]
  const float* window = (const float*)d_in[3];   // [400]
  float* out = (float*)d_out;                    // [16,200000]

  const int total = BB * OUTL;
  istft_zero_kernel<<<(total + 255) / 256, 256, 0, stream>>>(out, total);
  istft_wmma_kernel<<<BB * (TT / 16), 256, 0, stream>>>(mag, phase, weight, out);
  istft_norm_kernel<<<(total + 255) / 256, 256, 0, stream>>>(out, window);
}
